// QuestionAttention_41231686041950
// MI455X (gfx1250) — matrix operand order repaired and checked by reading, not compiled, not run on hardware
//
#include <hip/hip_runtime.h>
#include <hip/hip_bf16.h>
#include <math.h>
#include <stdint.h>

#define B   64
#define L   512
#define H2  1024

typedef __attribute__((ext_vector_type(16))) __bf16 v16bf;
typedef __attribute__((ext_vector_type(8)))  float  v8f;

// ---------------------------------------------------------------------------
// K0: Wt[n][k] = bf16(W_q[k][n])  -- one-time 4MB->2MB transpose+convert.
// ---------------------------------------------------------------------------
__global__ __launch_bounds__(256)
void wq_transpose_kernel(const float* __restrict__ W, __bf16* __restrict__ Wt) {
    __shared__ float tile[32][33];
    const int n0 = blockIdx.x * 32;
    const int k0 = blockIdx.y * 32;
    const int tx = threadIdx.x & 31;
    const int ty = threadIdx.x >> 5;       // 0..7
    for (int r = ty; r < 32; r += 8)
        tile[r][tx] = W[(size_t)(k0 + r) * H2 + (n0 + tx)];
    __syncthreads();
    for (int r = ty; r < 32; r += 8)
        Wt[(size_t)(n0 + r) * H2 + (k0 + tx)] = (__bf16)tile[tx][r];
}

// ---------------------------------------------------------------------------
// K1: dec_feat[b][n] = s_t_hat[b][:] @ W_qs_w[:][n] + W_qs_b[n]
// ---------------------------------------------------------------------------
__global__ __launch_bounds__(256)
void dec_feat_kernel(const float* __restrict__ s_t_hat,
                     const float* __restrict__ W,
                     const float* __restrict__ bias,
                     float* __restrict__ dec) {
    __shared__ float s_row[H2];
    const int b = blockIdx.x;
    const int n = blockIdx.y * 256 + threadIdx.x;
    for (int k = threadIdx.x; k < H2; k += 256)
        s_row[k] = s_t_hat[b * H2 + k];
    __syncthreads();
    float acc = bias[n];
    for (int k = 0; k < H2; ++k)
        acc = fmaf(s_row[k], W[(size_t)k * H2 + n], acc);
    dec[b * H2 + n] = acc;
}

// ---------------------------------------------------------------------------
// K2: fused scores. One WG (8 waves) per 32-row M-tile (2 M-subtiles).
// A tile staged via GLOBAL_LOAD_ASYNC_TO_LDS_B128 (ASYNCcnt DMA path),
// B fragments preloaded into a register array so the backend can clause
// 16 global_load_b128 and overlap them with the 16 WMMAs per K-step.
// ---------------------------------------------------------------------------
__global__ __launch_bounds__(256)
void scores_kernel(const float* __restrict__ pai_q,
                   const __bf16* __restrict__ Wt,
                   const float* __restrict__ dec,
                   const float* __restrict__ cov,
                   const float* __restrict__ W_c,
                   const float* __restrict__ v_q,
                   float* __restrict__ scores) {
    __shared__ float AshF[32 * 32];   // A tile, row-major 32 x 32 (f32)
    __shared__ float s_scores[32];

    const int tid     = threadIdx.x;
    const int wave    = tid >> 5;
    const int lane    = tid & 31;
    const int halfsel = lane >> 4;    // 0: lanes 0-15, 1: lanes 16-31
    const int lrow    = lane & 15;

    const int tileId = blockIdx.x;            // B * (L/32) tiles
    const int b      = tileId / (L / 32);
    const int l0     = (tileId % (L / 32)) * 32;
    const float* Abase = pai_q + ((size_t)b * L + l0) * H2;

    if (tid < 32) s_scores[tid] = 0.0f;

    // Per-thread async-copy slot: row r = tid>>3, 4 floats at col (tid&7)*4.
    const int ar = tid >> 3;
    const int ac = (tid & 7) * 4;
    const unsigned lds_dst  = (unsigned)(uintptr_t)&AshF[ar * 32 + ac];
    const unsigned goff_row = (unsigned)((ar * H2 + ac) * sizeof(float));
    const unsigned long long sbase = (unsigned long long)(uintptr_t)Abase;

    v8f acc[2][8];
    #pragma unroll
    for (int m = 0; m < 2; ++m)
        #pragma unroll
        for (int t = 0; t < 8; ++t) acc[m][t] = (v8f){0,0,0,0,0,0,0,0};

    for (int k0 = 0; k0 < H2; k0 += 32) {
        __syncthreads();                       // previous tile fully consumed
        // DMA 32x32 f32 A tile into LDS: 256 lanes x 16B, ASYNCcnt-tracked.
        {
            const unsigned goff = goff_row + (unsigned)(k0 * sizeof(float));
            asm volatile("global_load_async_to_lds_b128 %0, %1, %2"
                         :: "v"(lds_dst), "v"(goff), "s"(sbase) : "memory");
            asm volatile("s_wait_asynccnt 0x0" ::: "memory");
        }
        __syncthreads();

        // A fragments (contiguous k order, matched by B): M = lrow,
        // K(i) = i + 16*halfsel  -> one contiguous 16-f32 run
        v16bf afrag[2];
        #pragma unroll
        for (int m = 0; m < 2; ++m)
            #pragma unroll
            for (int i = 0; i < 16; ++i) {
                const int kk = i + (halfsel << 4);
                afrag[m][i] = (__bf16)AshF[(m * 16 + lrow) * 32 + kk];
            }

        // Preload all 8 B fragments (distinct regs -> clause + overlap).
        // B fragments, same k order as A: N = lrow, K(j) = j + 16*halfsel; contiguous in Wt.
        v16bf bfr[8];
        #pragma unroll
        for (int t = 0; t < 8; ++t) {
            const int n = wave * 128 + t * 16 + lrow;
            bfr[t] = *(const v16bf*)(Wt + (size_t)n * H2 + k0 + (halfsel << 4));
        }
        #pragma unroll
        for (int t = 0; t < 8; ++t) {
            acc[0][t] = __builtin_amdgcn_wmma_f32_16x16x32_bf16(
                false, afrag[0], false, bfr[t], (short)0, acc[0][t], false, false);
            acc[1][t] = __builtin_amdgcn_wmma_f32_16x16x32_bf16(
                false, afrag[1], false, bfr[t], (short)0, acc[1][t], false, false);
        }
    }

    // Epilogue. C tile layout: VGPR r holds M = r + 8*halfsel, N = lrow.
    const int mbase = halfsel * 8;
    float covv[2][8], partial[2][8];
    #pragma unroll
    for (int m = 0; m < 2; ++m)
        #pragma unroll
        for (int r = 0; r < 8; ++r) {
            covv[m][r]    = cov[b * L + l0 + m * 16 + mbase + r];
            partial[m][r] = 0.0f;
        }
    #pragma unroll
    for (int t = 0; t < 8; ++t) {
        const int n    = wave * 128 + t * 16 + lrow;
        const float d  = dec[b * H2 + n];
        const float wc = W_c[n];
        const float vq = v_q[n];
        #pragma unroll
        for (int m = 0; m < 2; ++m)
            #pragma unroll
            for (int r = 0; r < 8; ++r) {
                const float e = tanhf(acc[m][t][r] + d + covv[m][r] * wc);
                partial[m][r] = fmaf(e, vq, partial[m][r]);
            }
    }
    // reduce over the 16 lanes sharing each M row (xor masks < 16 stay in half)
    #pragma unroll
    for (int s = 8; s >= 1; s >>= 1)
        #pragma unroll
        for (int m = 0; m < 2; ++m)
            #pragma unroll
            for (int r = 0; r < 8; ++r)
                partial[m][r] += __shfl_xor(partial[m][r], s, 32);
    if (lrow == 0) {
        #pragma unroll
        for (int m = 0; m < 2; ++m)
            #pragma unroll
            for (int r = 0; r < 8; ++r)
                atomicAdd(&s_scores[m * 16 + mbase + r], partial[m][r]); // ds_add_f32
    }
    __syncthreads();
    if (tid < 32)
        scores[b * L + l0 + tid] = s_scores[tid];
}

// ---------------------------------------------------------------------------
// K3: masked softmax + renormalize + coverage update. One block per batch.
// ---------------------------------------------------------------------------
__global__ __launch_bounds__(L)
void softmax_kernel(const float* __restrict__ scores,
                    const float* __restrict__ mask,
                    const float* __restrict__ cov,
                    float* __restrict__ alpha_s,
                    float* __restrict__ out_alpha,
                    float* __restrict__ out_cov) {
    __shared__ float red[L / 32];
    __shared__ float bc;
    const int b = blockIdx.x, l = threadIdx.x;
    const int wave = l >> 5, lane = l & 31;

    float s = scores[b * L + l];

    float m = s;
    #pragma unroll
    for (int o = 16; o >= 1; o >>= 1) m = fmaxf(m, __shfl_xor(m, o, 32));
    if (lane == 0) red[wave] = m;
    __syncthreads();
    if (l == 0) {
        float t = red[0];
        for (int i = 1; i < L / 32; ++i) t = fmaxf(t, red[i]);
        bc = t;
    }
    __syncthreads();
    const float mx = bc;

    float e = __expf(s - mx);
    float se = e;
    #pragma unroll
    for (int o = 16; o >= 1; o >>= 1) se += __shfl_xor(se, o, 32);
    if (lane == 0) red[wave] = se;
    __syncthreads();
    if (l == 0) {
        float t = 0.f;
        for (int i = 0; i < L / 32; ++i) t += red[i];
        bc = t;
    }
    __syncthreads();
    const float sumE = bc;

    float a = (e / sumE) * mask[b * L + l];
    float sa = a;
    #pragma unroll
    for (int o = 16; o >= 1; o >>= 1) sa += __shfl_xor(sa, o, 32);
    if (lane == 0) red[wave] = sa;
    __syncthreads();
    if (l == 0) {
        float t = 0.f;
        for (int i = 0; i < L / 32; ++i) t += red[i];
        bc = t;
    }
    __syncthreads();
    const float alpha = a / bc;

    alpha_s  [b * L + l] = alpha;
    out_alpha[b * L + l] = alpha;
    out_cov  [b * L + l] = cov[b * L + l] + alpha;
}

// ---------------------------------------------------------------------------
// K4: c_q_t[b][h] = sum_l alpha[b][l] * pai_q[b][l][h]  (bandwidth bound)
// ---------------------------------------------------------------------------
__global__ __launch_bounds__(128)
void context_kernel(const float* __restrict__ pai_q,
                    const float* __restrict__ alpha,
                    float* __restrict__ out_c) {
    __shared__ float a_sh[L];
    const int b = blockIdx.x;
    const int h = blockIdx.y * 128 + threadIdx.x;
    for (int l = threadIdx.x; l < L; l += 128)
        a_sh[l] = alpha[b * L + l];
    __syncthreads();
    const float* base = pai_q + (size_t)b * L * H2 + h;
    float acc = 0.0f;
    #pragma unroll 4
    for (int l = 0; l < L; ++l)
        acc = fmaf(a_sh[l], base[(size_t)l * H2], acc);
    out_c[b * H2 + h] = acc;
}

// ---------------------------------------------------------------------------
extern "C" void kernel_launch(void* const* d_in, const int* in_sizes, int n_in,
                              void* d_out, int out_size, void* d_ws, size_t ws_size,
                              hipStream_t stream) {
    const float* s_t_hat = (const float*)d_in[0];   // (B, H2)
    const float* pai_q   = (const float*)d_in[1];   // (B, L, H2)
    const float* maskp   = (const float*)d_in[2];   // (B, L)
    const float* covp    = (const float*)d_in[3];   // (B, L)
    const float* W_q     = (const float*)d_in[4];   // (H2, H2)
    const float* W_qs_w  = (const float*)d_in[5];   // (H2, H2)
    const float* W_qs_b  = (const float*)d_in[6];   // (H2,)
    const float* W_c     = (const float*)d_in[7];   // (1, H2)
    const float* v_q     = (const float*)d_in[8];   // (H2, 1)

    float* out       = (float*)d_out;
    float* out_c     = out;                         // (B, H2)
    float* out_alpha = out + B * H2;                // (B, L)
    float* out_cov   = out + B * H2 + B * L;        // (B, L)

    // Workspace layout (Wt first for 32B alignment of v16bf loads):
    __bf16* Wt      = (__bf16*)d_ws;
    float* dec_feat = (float*)((char*)d_ws + (size_t)H2 * H2 * sizeof(__bf16));
    float* scoresp  = dec_feat + B * H2;
    float* alpha_s  = scoresp + B * L;

    wq_transpose_kernel<<<dim3(H2 / 32, H2 / 32), 256, 0, stream>>>(W_q, Wt);

    dec_feat_kernel<<<dim3(B, H2 / 256), 256, 0, stream>>>(
        s_t_hat, W_qs_w, W_qs_b, dec_feat);

    scores_kernel<<<B * (L / 32), 256, 0, stream>>>(
        pai_q, Wt, dec_feat, covp, W_c, v_q, scoresp);

    softmax_kernel<<<B, L, 0, stream>>>(
        scoresp, maskp, covp, alpha_s, out_alpha, out_cov);

    context_kernel<<<dim3(B, H2 / 128), 128, 0, stream>>>(
        pai_q, alpha_s, out_c);
}
